// QLSTM_65481071407279
// MI455X (gfx1250) — matrix operand order repaired and checked by reading, not compiled, not run on hardware
//
#include <hip/hip_runtime.h>
#include <hip/hip_bf16.h>

typedef __attribute__((ext_vector_type(16))) __bf16 v16bf;
typedef __attribute__((ext_vector_type(8)))  __bf16 v8bf;
typedef __attribute__((ext_vector_type(8)))  float  v8f;

#define T_STEPS 2048
#define BATCH   64
#define DIN     256
#define HID     256
#define KDIM    512     // D + H
#define NWG     16      // persistent workgroups; each owns HID/NWG hidden units
#define HS      16      // hidden units per WG
#define NCOLS   64      // 4 gates * HS
#define NTHREADS 256    // 8 wave32s

__device__ __forceinline__ unsigned short f2bf(float f) {
  unsigned u = __float_as_uint(f);
  unsigned r = u + 0x7fffu + ((u >> 16) & 1u);   // round-to-nearest-even
  return (unsigned short)(r >> 16);
}
__device__ __forceinline__ float sigmoidf_(float x) { return 1.0f / (1.0f + __expf(-x)); }

union AFrag { v16bf v; v8bf h[2]; };

__global__ void lstm_init(float* hbuf0, unsigned* cnt) {
  int i = blockIdx.x * blockDim.x + threadIdx.x;
  if (i < BATCH * HID) hbuf0[i] = 0.0f;
  if (i == 0) *cnt = 0u;
}

__global__ __launch_bounds__(NTHREADS, 1)
void lstm_persistent(const float* __restrict__ x,
                     const float* __restrict__ Wf, const float* __restrict__ bfv,
                     const float* __restrict__ Wi, const float* __restrict__ biv,
                     const float* __restrict__ Wg, const float* __restrict__ bgv,
                     const float* __restrict__ Wo, const float* __restrict__ bov,
                     float* __restrict__ out,
                     float* __restrict__ hbuf0, float* __restrict__ hbuf1,
                     unsigned* __restrict__ cnt) {
  // LDS: 64K swizzled bf16 weights + 64K bf16 comb + 16K gates + 4K cell + bias
  __shared__ __align__(64) unsigned short wS[4 * 16 * 32 * 16];
  __shared__ __align__(64) unsigned short combS[BATCH * KDIM];
  __shared__ float gateS[4 * BATCH * HS];
  __shared__ float cS[BATCH * HS];
  __shared__ float biasS[NCOLS];

  const int tid  = threadIdx.x;
  const int wg   = blockIdx.x;       // hidden slice [wg*HS, wg*HS+HS)
  const int lane = tid & 31;
  const int wid  = tid >> 5;         // 0..7
  const int hl   = lane >> 4;        // half-wave
  const int nn   = lane & 15;
  const int mt   = wid >> 1;         // M tile 0..3 (rows 16*mt..)
  const int nt0  = (wid & 1) * 2;    // gate tiles: {0,1} or {2,3}
  const int nt1  = nt0 + 1;

  // ---- one-time: stage+swizzle weight slice (fp32 global -> bf16 LDS, B-fragment order)
  // wS element ((nt*16+ks)*32+ln)*16+e  holds  W_gate[k = 32*ks + 16*(e>>3) + 8*(ln>>4) + (e&7)][wg*HS + (ln&15)]
  for (int j = 0; j < (4 * 16 * 32 * 16) / NTHREADS; ++j) {
    int f  = tid + j * NTHREADS;
    int e  = f & 15, ln = (f >> 4) & 31, ks = (f >> 9) & 15, nt = f >> 13;
    int k  = ks * 32 + (e >> 3) * 16 + (ln >> 4) * 8 + (e & 7);
    int col = wg * HS + (ln & 15);
    const float* Ws = (nt == 0) ? Wf : (nt == 1) ? Wi : (nt == 2) ? Wg : Wo;
    wS[f] = f2bf(Ws[k * HID + col]);
  }
  if (tid < NCOLS) {
    int nt = tid >> 4;
    const float* bs = (nt == 0) ? bfv : (nt == 1) ? biv : (nt == 2) ? bgv : bov;
    biasS[tid] = bs[wg * HS + (tid & 15)];
  }
  for (int p = tid; p < BATCH * HS; p += NTHREADS) cS[p] = 0.0f;  // c0 = 0
  __syncthreads();

  const int arow = mt * 16 + nn;  // A-fragment row for this lane

  for (int t = 0; t < T_STEPS; ++t) {
    const float* hin  = (t & 1) ? hbuf1 : hbuf0;
    float*       hout = (t & 1) ? hbuf0 : hbuf1;
    const float* xt   = x + (size_t)t * (BATCH * DIN);

    // ---- build comb = [x_t | h_{t-1}] as bf16 in LDS (coalesced, k fastest)
    for (int j = 0; j < (BATCH * KDIM) / NTHREADS; ++j) {
      int i = tid + j * NTHREADS;
      int b = i >> 9, k = i & 511;
      float v = (k < DIN) ? xt[b * DIN + k] : hin[b * HID + (k - DIN)];
      combS[i] = f2bf(v);
    }
    if (t + 1 < T_STEPS) __builtin_prefetch(xt + BATCH * DIN + tid * 64, 0, 1);
    __syncthreads();

    // ---- GEMM: [64 x 512] @ [512 x 64] via v_wmma_f32_16x16x32_bf16, 2 N-tiles/wave
    v8f acc0 = {};
    v8f acc1 = {};
    #pragma unroll
    for (int ks = 0; ks < 16; ++ks) {
      const int kb = ks * 32;
      AFrag a;  // 16-bit A 16x32 striping: lanes 0-15 K 0-7/16-23, lanes 16-31 K 8-15/24-31
      a.h[0] = *reinterpret_cast<const v8bf*>(&combS[arow * KDIM + kb + 8 * hl]);
      a.h[1] = *reinterpret_cast<const v8bf*>(&combS[arow * KDIM + kb + 16 + 8 * hl]);
      v16bf b0 = *reinterpret_cast<const v16bf*>(&wS[((nt0 * 16 + ks) * 32 + lane) * 16]);
      v16bf b1 = *reinterpret_cast<const v16bf*>(&wS[((nt1 * 16 + ks) * 32 + lane) * 16]);
      acc0 = __builtin_amdgcn_wmma_f32_16x16x32_bf16(false, a.v, false, b0, (short)0, acc0, false, false);
      acc1 = __builtin_amdgcn_wmma_f32_16x16x32_bf16(false, a.v, false, b1, (short)0, acc1, false, false);
    }

    // ---- bias + activation in registers, stage gates in LDS
    // D layout: lane = n + 16*(m>=8), vgpr r = m%8  ->  m = 16*mt + 8*hl + r, n = nn
    #pragma unroll
    for (int r = 0; r < 8; ++r) {
      int m = mt * 16 + hl * 8 + r;
      float v0 = acc0[r] + biasS[nt0 * 16 + nn];
      float v1 = acc1[r] + biasS[nt1 * 16 + nn];
      v0 = (nt0 == 2) ? tanhf(v0) : sigmoidf_(v0);   // gate 2 = g -> tanh, else sigmoid
      v1 = (nt1 == 2) ? tanhf(v1) : sigmoidf_(v1);
      gateS[(nt0 * BATCH + m) * HS + nn] = v0;
      gateS[(nt1 * BATCH + m) * HS + nn] = v1;
    }
    __syncthreads();

    // ---- LSTM cell update for this WG's hidden slice (c stays in LDS)
    for (int p = tid; p < BATCH * HS; p += NTHREADS) {
      int b = p >> 4, jj = p & 15;
      float fg = gateS[(0 * BATCH + b) * HS + jj];
      float ig = gateS[(1 * BATCH + b) * HS + jj];
      float gg = gateS[(2 * BATCH + b) * HS + jj];
      float og = gateS[(3 * BATCH + b) * HS + jj];
      float cn = fg * cS[p] + ig * gg;
      float hn = og * tanhf(cn);
      cS[p] = cn;
      int col = wg * HS + jj;
      hout[b * HID + col] = hn;
      out[(size_t)t * (BATCH * HID) + b * HID + col] = hn;
      if (t == T_STEPS - 1) {
        out[(size_t)T_STEPS * (BATCH * HID) + b * HID + col] = hn;                // hx
        out[(size_t)T_STEPS * (BATCH * HID) + BATCH * HID + b * HID + col] = cn;  // cx
      }
    }

    // ---- grid-wide step barrier (16 persistent WGs, monotonic counter)
    __syncthreads();
    if (tid == 0) {
      __threadfence();  // release h writes
      __hip_atomic_fetch_add(cnt, 1u, __ATOMIC_ACQ_REL, __HIP_MEMORY_SCOPE_AGENT);
      const unsigned target = (unsigned)(t + 1) * NWG;
      while (__hip_atomic_load(cnt, __ATOMIC_ACQUIRE, __HIP_MEMORY_SCOPE_AGENT) < target)
        __builtin_amdgcn_s_sleep(2);
    }
    __syncthreads();
    __threadfence();  // acquire: make other WGs' h visible to all waves
  }
}

extern "C" void kernel_launch(void* const* d_in, const int* in_sizes, int n_in,
                              void* d_out, int out_size, void* d_ws, size_t ws_size,
                              hipStream_t stream) {
  const float* x  = (const float*)d_in[0];
  const float* Wf = (const float*)d_in[1];
  const float* bf = (const float*)d_in[2];
  const float* Wi = (const float*)d_in[3];
  const float* bi = (const float*)d_in[4];
  const float* Wg = (const float*)d_in[5];
  const float* bg = (const float*)d_in[6];
  const float* Wo = (const float*)d_in[7];
  const float* bo = (const float*)d_in[8];
  float* out = (float*)d_out;

  float*    hbuf0 = (float*)d_ws;                 // [64*256] h ping
  float*    hbuf1 = hbuf0 + BATCH * HID;          // [64*256] h pong
  unsigned* cnt   = (unsigned*)(hbuf1 + BATCH * HID);

  lstm_init<<<dim3((BATCH * HID + 255) / 256), dim3(256), 0, stream>>>(hbuf0, cnt);
  lstm_persistent<<<dim3(NWG), dim3(NTHREADS), 0, stream>>>(
      x, Wf, bf, Wi, bi, Wg, bg, Wo, bo, out, hbuf0, hbuf1, cnt);
}
